// KnnPointTransformer_23922967838812
// MI455X (gfx1250) — compile-verified, matrix operand order checked by reading, not run on hardware
//
#include <hip/hip_runtime.h>
#include <math.h>

#define BATCH 2
#define NPTS  8192
#define DM    128
#define KNN   16
#define TOTP  (BATCH * NPTS)   // 16384 points

#define WSTR  136   // padded f16 LDS row stride (272 B = 68 dwords, conflict-free)
#define FSTR  132   // padded f32 LDS row stride (528 B = 132 dwords, conflict-free)

typedef __attribute__((ext_vector_type(16))) _Float16 v16h;
typedef __attribute__((ext_vector_type(8)))  float    v8f;

__device__ __forceinline__ v8f vzero8() {
    v8f z;
#pragma unroll
    for (int i = 0; i < 8; ++i) z[i] = 0.0f;
    return z;
}

// 8-arg WMMA: (neg_a, A, neg_b, B, c_mod, C, reuse_a, reuse_b)
__device__ __forceinline__ v8f wmma16(v16h a, v16h b, v8f c) {
    return __builtin_amdgcn_wmma_f32_16x16x32_f16(false, a, false, b, (short)0, c,
                                                  false, false);
}

// Load a 16-element f16 fragment for row `row` of a row-major matrix with a
// given row stride (in halfs).  A-matrix 16x32 layout (wave32): lane group
// hi=0 -> K=k0+0..7 and k0+16..23; hi=1 -> K=k0+8..15 and k0+24..31.  The same
// formula serves B = W^T fragments because B[k][col] = W[col][k] (row := col).
__device__ __forceinline__ v16h load_frag(const _Float16* base, int row, int k0,
                                          int hi, int stride) {
    const _Float16* pt = base + row * stride + k0 + hi * 8;
    v16h a;
#pragma unroll
    for (int j = 0; j < 8; ++j) {
        a[j]     = pt[j];
        a[8 + j] = pt[16 + j];
    }
    return a;
}

// Async Global->LDS copy of one 16-byte packet per lane (CDNA5 ASYNCcnt path).
__device__ __forceinline__ void async_copy_b128(const void* gsrc, void* lds_dst) {
    const uint32_t ldsa = (uint32_t)(uintptr_t)lds_dst;   // low 32 bits = LDS offset
    asm volatile("global_load_async_to_lds_b128 %0, %1, off"
                 :: "v"(ldsa), "v"(gsrc)
                 : "memory");
}

__device__ __forceinline__ void async_wait0() {
    asm volatile("s_wait_asynccnt 0x0" ::: "memory");
}

// Stage a 128x128 f16 matrix from global (row stride DM) into LDS with padded
// row stride WSTR, using async b128 packets (16 packets per row).
__device__ __forceinline__ void stage_weight(const _Float16* __restrict__ g,
                                             _Float16* lds, int tid, int nthr) {
    for (int i = tid; i < DM * (DM / 8); i += nthr) {   // 2048 packets
        const int row = i >> 4, seg = i & 15;
        async_copy_b128(g + row * DM + seg * 8, lds + row * WSTR + seg * 8);
    }
}

// ---------------------------------------------------------------------------
// K0: fp32 -> fp16 weight conversion
// ---------------------------------------------------------------------------
__global__ void cvt_f16_kernel(const float* __restrict__ src,
                               _Float16* __restrict__ dst, int nel) {
    int i = blockIdx.x * blockDim.x + threadIdx.x;
    if (i < nel) dst[i] = (_Float16)src[i];
}

// ---------------------------------------------------------------------------
// K1: KNN (16 nearest incl. self).  Thread per query, LDS-tiled candidates,
// fully unrolled register insertion sort (list stays in VGPRs).
// ---------------------------------------------------------------------------
__global__ void __launch_bounds__(256) knn_kernel(const float* __restrict__ xyz,
                                                  int* __restrict__ knn) {
    __shared__ float cx[256], cy[256], cz[256];
    const int tid = threadIdx.x;
    const int p   = blockIdx.x * 256 + tid;
    const int b   = p >> 13;       // p / 8192
    const int n   = p & (NPTS - 1);
    const float* xb = xyz + b * 3 * NPTS;
    const float qx = xb[n], qy = xb[NPTS + n], qz = xb[2 * NPTS + n];

    float bd[KNN];
    int   bi[KNN];
#pragma unroll
    for (int k = 0; k < KNN; ++k) { bd[k] = 3.0e38f; bi[k] = 0; }

    for (int j0 = 0; j0 < NPTS; j0 += 256) {
        __syncthreads();
        cx[tid] = xb[j0 + tid];
        cy[tid] = xb[NPTS + j0 + tid];
        cz[tid] = xb[2 * NPTS + j0 + tid];
        __syncthreads();
        for (int t = 0; t < 256; ++t) {
            const float dx = qx - cx[t], dy = qy - cy[t], dz = qz - cz[t];
            float d = dx * dx + dy * dy + dz * dz;
            if (d < bd[KNN - 1]) {
                int jj = j0 + t;
#pragma unroll
                for (int s = 0; s < KNN; ++s) {
                    const bool sw = d < bd[s];
                    const float td = sw ? bd[s] : d;
                    const int   ti = sw ? bi[s] : jj;
                    bd[s] = sw ? d  : bd[s];
                    bi[s] = sw ? jj : bi[s];
                    d = td; jj = ti;
                }
            }
        }
    }
#pragma unroll
    for (int k = 0; k < KNN; ++k) knn[p * KNN + k] = bi[k];
}

// ---------------------------------------------------------------------------
// K2a: x = feat @ fc1_w.T + fc1_b  (3 -> 128), stored f16 for the WMMA GEMMs.
// ---------------------------------------------------------------------------
__global__ void fc1_kernel(const float* __restrict__ features,
                           const float* __restrict__ fc1_w,
                           const float* __restrict__ fc1_b,
                           _Float16* __restrict__ xh) {
    const int p = blockIdx.x;
    const int c = threadIdx.x;          // 128
    const int b = p >> 13, n = p & (NPTS - 1);
    const float* fb = features + b * 3 * NPTS;
    const float f0 = fb[n], f1 = fb[NPTS + n], f2 = fb[2 * NPTS + n];
    const float v = f0 * fc1_w[c * 3 + 0] + f1 * fc1_w[c * 3 + 1] +
                    f2 * fc1_w[c * 3 + 2] + fc1_b[c];
    xh[p * DM + c] = (_Float16)v;
}

// ---------------------------------------------------------------------------
// K2b: q/k/v = x @ W.T via WMMA.  Weight matrix staged into padded LDS via
// async Global->LDS once per block; per K-chunk all 8 B-fragments preloaded
// so the 8 WMMAs issue back-to-back.
// ---------------------------------------------------------------------------
__global__ void __launch_bounds__(256) qkv_kernel(const _Float16* __restrict__ xh,
                                                  const _Float16* __restrict__ whq,
                                                  const _Float16* __restrict__ whk,
                                                  const _Float16* __restrict__ whv,
                                                  float* __restrict__ outq,
                                                  float* __restrict__ outk,
                                                  float* __restrict__ outv) {
    extern __shared__ char smem_raw[];
    _Float16* wsh = (_Float16*)smem_raw;            // 128 x WSTR f16 = 34 KB

    const int tid = threadIdx.x;
    const _Float16* wh = (blockIdx.y == 0) ? whq : (blockIdx.y == 1) ? whk : whv;
    float* out = (blockIdx.y == 0) ? outq : (blockIdx.y == 1) ? outk : outv;

    stage_weight(wh, wsh, tid, 256);
    async_wait0();
    __syncthreads();

    const int w = tid >> 5, lane = tid & 31;
    const int m = lane & 15, hi = lane >> 4;
    const int mtile = blockIdx.x * 8 + w;           // 16 rows each, 1024 tiles

    v8f acc[8];
#pragma unroll
    for (int t = 0; t < 8; ++t) acc[t] = vzero8();

#pragma unroll
    for (int c = 0; c < 4; ++c) {
        const int k0 = c * 32;
        const v16h a = load_frag(xh, mtile * 16 + m, k0, hi, DM);
        v16h bf[8];
#pragma unroll
        for (int t = 0; t < 8; ++t)
            bf[t] = load_frag(wsh, t * 16 + m, k0, hi, WSTR);
#pragma unroll
        for (int t = 0; t < 8; ++t)
            acc[t] = wmma16(a, bf[t], acc[t]);
    }
#pragma unroll
    for (int t = 0; t < 8; ++t)
#pragma unroll
        for (int r = 0; r < 8; ++r)
            out[(size_t)(mtile * 16 + r + 8 * hi) * DM + t * 16 + m] = acc[t][r];
}

// ---------------------------------------------------------------------------
// K3: fused per-point attention.  One wave per point (its K=16 neighbors are
// the WMMA M dimension).  pos MLP layer2, gamma layer1, gamma layer2 all via
// v_wmma_f32_16x16x32_f16 against LDS-resident padded f16 weights (102 KB,
// async-staged), with per-wave padded LDS staging for the layer1->layer2
// transpose, pos, and gamma.  8 waves/block, 268 KB LDS of the 320 KB WGP.
// ---------------------------------------------------------------------------
__global__ void __launch_bounds__(256) attn_kernel(
    const float* __restrict__ q, const float* __restrict__ kx,
    const float* __restrict__ vx, const int* __restrict__ knn,
    const float* __restrict__ xyz,
    const float* __restrict__ d1_w, const float* __restrict__ d1_b,
    const _Float16* __restrict__ d2_h, const float* __restrict__ d2_b,
    const _Float16* __restrict__ g1_h, const float* __restrict__ g1_b,
    const _Float16* __restrict__ g2_h, const float* __restrict__ g2_b,
    float* __restrict__ res) {
    extern __shared__ char smem_raw[];
    _Float16* wd2 = (_Float16*)smem_raw;                 // 128*WSTR f16 = 34 KB
    _Float16* wg1 = wd2 + DM * WSTR;                     // 34 KB
    _Float16* wg2 = wg1 + DM * WSTR;                     // 34 KB
    float*    posf = (float*)(wg2 + DM * WSTR);          // 8 * 16*FSTR f32 = 66 KB
    _Float16* stg  = (_Float16*)(posf + 8 * 16 * FSTR); // 8 * 16*WSTR f16 = 34 KB
    float*    gamf = (float*)(stg + 8 * 16 * WSTR);      // 8 * 16*FSTR f32 = 66 KB

    const int tid = threadIdx.x;
    // Async weight staging into padded LDS: 3 x 2048 b128 packets.
    stage_weight(d2_h, wd2, tid, 256);
    stage_weight(g1_h, wg1, tid, 256);
    stage_weight(g2_h, wg2, tid, 256);

    const int w = tid >> 5, lane = tid & 31;
    const int m = lane & 15, hi = lane >> 4;
    const int p = blockIdx.x * 8 + w;               // point id
    const int b = p >> 13, n = p & (NPTS - 1);

    float* posw = posf + w * 16 * FSTR;
    _Float16* stw = stg + w * 16 * WSTR;
    float* gamw = gamf + w * 16 * FSTR;

    // KNN ids up front; prefetch gathered v rows into cache for the tail.
    int ids[KNN];
#pragma unroll
    for (int k = 0; k < KNN; ++k) ids[k] = knn[p * KNN + k];
#pragma unroll
    for (int k = 0; k < KNN; ++k)
        __builtin_prefetch(vx + ((size_t)(b * NPTS + ids[k]) * DM) + lane * 4, 0, 0);

    const float* xb = xyz + b * 3 * NPTS;
    const float qx = xb[n], qy = xb[NPTS + n], qz = xb[2 * NPTS + n];
    const int idm = ids[m];                         // neighbor of row m
    const float rx = qx - xb[idm];
    const float ry = qy - xb[NPTS + idm];
    const float rz = qz - xb[2 * NPTS + idm];

    async_wait0();
    __syncthreads();

    // ---- pos MLP layer 1 (3 -> 128), relu, to f16 staging -----------------
    for (int cc = 0; cc < 64; ++cc) {
        const int c = hi * 64 + cc;
        const float t = rx * d1_w[c * 3 + 0] + ry * d1_w[c * 3 + 1] +
                        rz * d1_w[c * 3 + 2] + d1_b[c];
        stw[m * WSTR + c] = (_Float16)fmaxf(t, 0.0f);
    }
    __syncthreads();

    v8f acc[8];
    // ---- pos MLP layer 2: (16x128) @ d2.T via WMMA ------------------------
#pragma unroll
    for (int t = 0; t < 8; ++t) acc[t] = vzero8();
#pragma unroll
    for (int c = 0; c < 4; ++c) {
        const int k0 = c * 32;
        const v16h a = load_frag(stw, m, k0, hi, WSTR);
        v16h bf[8];
#pragma unroll
        for (int t = 0; t < 8; ++t) bf[t] = load_frag(wd2, t * 16 + m, k0, hi, WSTR);
#pragma unroll
        for (int t = 0; t < 8; ++t) acc[t] = wmma16(a, bf[t], acc[t]);
    }
#pragma unroll
    for (int t = 0; t < 8; ++t)
#pragma unroll
        for (int r = 0; r < 8; ++r) {
            const int col = t * 16 + m, row = r + 8 * hi;
            posw[row * FSTR + col] = acc[t][r] + d2_b[col];
        }
    __syncthreads();

    // ---- gamma layer 1: h = q - k_g + pos, then (16x128)@g1.T -------------
    const float* qrow = q + (size_t)p * DM;
    const float* krow = kx + (size_t)(b * NPTS + idm) * DM;
#pragma unroll
    for (int t = 0; t < 8; ++t) acc[t] = vzero8();
#pragma unroll
    for (int c = 0; c < 4; ++c) {
        const int k0 = c * 32;
        v16h a;
#pragma unroll
        for (int j = 0; j < 8; ++j) {
            const int c0 = k0 + hi * 8 + j;
            a[j]     = (_Float16)(qrow[c0]      - krow[c0]      + posw[m * FSTR + c0]);
            a[8 + j] = (_Float16)(qrow[c0 + 16] - krow[c0 + 16] + posw[m * FSTR + c0 + 16]);
        }
        v16h bf[8];
#pragma unroll
        for (int t = 0; t < 8; ++t) bf[t] = load_frag(wg1, t * 16 + m, k0, hi, WSTR);
#pragma unroll
        for (int t = 0; t < 8; ++t) acc[t] = wmma16(a, bf[t], acc[t]);
    }
    __syncthreads();
#pragma unroll
    for (int t = 0; t < 8; ++t)
#pragma unroll
        for (int r = 0; r < 8; ++r) {
            const int col = t * 16 + m, row = r + 8 * hi;
            stw[row * WSTR + col] = (_Float16)fmaxf(acc[t][r] + g1_b[col], 0.0f);
        }
    __syncthreads();

    // ---- gamma layer 2 ----------------------------------------------------
#pragma unroll
    for (int t = 0; t < 8; ++t) acc[t] = vzero8();
#pragma unroll
    for (int c = 0; c < 4; ++c) {
        const int k0 = c * 32;
        const v16h a = load_frag(stw, m, k0, hi, WSTR);
        v16h bf[8];
#pragma unroll
        for (int t = 0; t < 8; ++t) bf[t] = load_frag(wg2, t * 16 + m, k0, hi, WSTR);
#pragma unroll
        for (int t = 0; t < 8; ++t) acc[t] = wmma16(a, bf[t], acc[t]);
    }
#pragma unroll
    for (int t = 0; t < 8; ++t)
#pragma unroll
        for (int r = 0; r < 8; ++r) {
            const int col = t * 16 + m, row = r + 8 * hi;
            gamw[row * FSTR + col] = acc[t][r] + g2_b[col];
        }
    __syncthreads();

    // ---- softmax over K (16 rows) + weighted sum of (v_g + pos) -----------
    const float scale = 0.0883883476483184f;       // 1/sqrt(128)
    for (int cc = 0; cc < 4; ++cc) {
        const int col = lane * 4 + cc;
        float g[KNN];
        float mx = -3.0e38f;
#pragma unroll
        for (int k = 0; k < KNN; ++k) {
            g[k] = gamw[k * FSTR + col] * scale;
            mx = fmaxf(mx, g[k]);
        }
        float s = 0.0f;
#pragma unroll
        for (int k = 0; k < KNN; ++k) { g[k] = __expf(g[k] - mx); s += g[k]; }
        const float inv = 1.0f / s;
        float r = 0.0f;
#pragma unroll
        for (int k = 0; k < KNN; ++k)
            r += g[k] * inv *
                 (vx[(size_t)(b * NPTS + ids[k]) * DM + col] + posw[k * FSTR + col]);
        res[(size_t)p * DM + col] = r;
    }
}

// ---------------------------------------------------------------------------
// K4: out = res @ fc2.T + fc2_b + features  (128 -> 3), transposed to (B,3,N)
// ---------------------------------------------------------------------------
__global__ void fc2_kernel(const float* __restrict__ res,
                           const float* __restrict__ fc2_w,
                           const float* __restrict__ fc2_b,
                           const float* __restrict__ features,
                           float* __restrict__ out) {
    const int p = blockIdx.x * blockDim.x + threadIdx.x;
    const int b = p >> 13, n = p & (NPTS - 1);
    const float* rr = res + (size_t)p * DM;
#pragma unroll
    for (int d = 0; d < 3; ++d) {
        float s = fc2_b[d];
        for (int c = 0; c < DM; ++c) s += rr[c] * fc2_w[d * DM + c];
        out[b * 3 * NPTS + d * NPTS + n] = s + features[b * 3 * NPTS + d * NPTS + n];
    }
}

// ---------------------------------------------------------------------------
extern "C" void kernel_launch(void* const* d_in, const int* in_sizes, int n_in,
                              void* d_out, int out_size, void* d_ws, size_t ws_size,
                              hipStream_t stream) {
    const float* xyz      = (const float*)d_in[0];
    const float* features = (const float*)d_in[1];
    const float* fc1_w = (const float*)d_in[2];
    const float* fc1_b = (const float*)d_in[3];
    const float* fc2_w = (const float*)d_in[4];
    const float* fc2_b = (const float*)d_in[5];
    const float* d1_w  = (const float*)d_in[6];
    const float* d1_b  = (const float*)d_in[7];
    const float* d2_w  = (const float*)d_in[8];
    const float* d2_b  = (const float*)d_in[9];
    const float* g1_w  = (const float*)d_in[10];
    const float* g1_b  = (const float*)d_in[11];
    const float* g2_w  = (const float*)d_in[12];
    const float* g2_b  = (const float*)d_in[13];
    const float* wq_w  = (const float*)d_in[14];
    const float* wk_w  = (const float*)d_in[15];
    const float* wv_w  = (const float*)d_in[16];
    float* out = (float*)d_out;

    // Workspace carve-up
    char* ws = (char*)d_ws;
    int* knn = (int*)ws;                 ws += (size_t)TOTP * KNN * 4;     // 1 MB
    _Float16* xh = (_Float16*)ws;        ws += (size_t)TOTP * DM * 2;      // 4 MB
    float* q  = (float*)ws;              ws += (size_t)TOTP * DM * 4;      // 8 MB
    float* kx = (float*)ws;              ws += (size_t)TOTP * DM * 4;      // 8 MB
    float* vx = (float*)ws;              ws += (size_t)TOTP * DM * 4;      // 8 MB
    float* res = (float*)ws;             ws += (size_t)TOTP * DM * 4;      // 8 MB
    _Float16* d2h = (_Float16*)ws;       ws += DM * DM * 2;
    _Float16* g1h = (_Float16*)ws;       ws += DM * DM * 2;
    _Float16* g2h = (_Float16*)ws;       ws += DM * DM * 2;
    _Float16* wqh = (_Float16*)ws;       ws += DM * DM * 2;
    _Float16* wkh = (_Float16*)ws;       ws += DM * DM * 2;
    _Float16* wvh = (_Float16*)ws;       ws += DM * DM * 2;

    const int nel = DM * DM;
    const int cb = (nel + 255) / 256;
    cvt_f16_kernel<<<cb, 256, 0, stream>>>(d2_w, d2h, nel);
    cvt_f16_kernel<<<cb, 256, 0, stream>>>(g1_w, g1h, nel);
    cvt_f16_kernel<<<cb, 256, 0, stream>>>(g2_w, g2h, nel);
    cvt_f16_kernel<<<cb, 256, 0, stream>>>(wq_w, wqh, nel);
    cvt_f16_kernel<<<cb, 256, 0, stream>>>(wk_w, wkh, nel);
    cvt_f16_kernel<<<cb, 256, 0, stream>>>(wv_w, wvh, nel);

    knn_kernel<<<TOTP / 256, 256, 0, stream>>>(xyz, knn);
    fc1_kernel<<<TOTP, DM, 0, stream>>>(features, fc1_w, fc1_b, xh);
    qkv_kernel<<<dim3(TOTP / 16 / 8, 3), 256, (size_t)DM * WSTR * 2, stream>>>(
        xh, wqh, wkh, wvh, q, kx, vx);
    const size_t smem = (size_t)(3 * DM * WSTR * 2)            // weights  104448
                      + (size_t)(8 * 16 * FSTR * 4)            // pos       67584
                      + (size_t)(8 * 16 * WSTR * 2)            // stage     34816
                      + (size_t)(8 * 16 * FSTR * 4);           // gamma     67584
    attn_kernel<<<TOTP / 8, 256, smem, stream>>>(q, kx, vx, knn, xyz,
                                                 d1_w, d1_b, d2h, d2_b,
                                                 g1h, g1_b, g2h, g2_b, res);
    fc2_kernel<<<TOTP / 128, 128, 0, stream>>>(res, fc2_w, fc2_b, features, out);
}
